// UpdateLayer_37134287242020
// MI455X (gfx1250) — matrix operand order repaired and checked by reading, not compiled, not run on hardware
//
#include <hip/hip_runtime.h>
#include <cmath>

// ---------------------------------------------------------------------------
// MI455X (gfx1250) implementation of the equivariant GNN forward pass.
//
// Input pointer map (depth-first, insertion-order flattening of setup_inputs):
//   d_in[0]  atom_features   (16000 x 192) f32
//   d_in[1]  atom_pos        (16000 x 3)   f32
//   d_in[2]  bond_features   (34000 x 64)  f32
//   d_in[3]  bond_sh         (34000 x 9)   f32
//   d_in[4]  bond_edge_index (2 x 34000)   i32
//   d_in[5]  spatial_features(256000 x 64) f32
//   d_in[6]  spatial_sh      (256000 x 9)  f32
//   d_in[7]  spatial_edge_idx(2 x 256000)  i32
//   d_in[8]  rotatable_bonds (1200)        i32
//   d_in[9]  batch           (unused)
//   d_in[10] update_instructs(16000)       i32
//   d_in[11] tor_edge_index  (2 x 28800)   i32
//   d_in[12..] params, depth-first insertion order:
//     bond[i] (i=0..4), 16 leaves each:
//       fc_w1,fc_b1,fc_w2,fc_b2,bn_gs,bn_bs,bn_gv,
//       eu_lin,eu_w1,eu_b1,eu_w2,eu_b2,eu_w3,eu_b3,eu_ln_g,eu_ln_b
//     spatial[i] (i=0..4), 7 leaves each: fc_w1,fc_b1,fc_w2,fc_b2,bn_gs,bn_bs,bn_gv
//     emb_w1, emb_b1, emb_w2, emb_b2
//     tor: fc_w1,fc_b1,fc_w2,fc_b2,bn_g,bn_b,fin_w1,fin_w2
// Output: new_pos(48000) | atom_features(3072000) | bond_features(2176000) | tor(1200)
// ---------------------------------------------------------------------------

typedef _Float16 f16;
typedef __attribute__((ext_vector_type(16))) _Float16 v16h;
typedef __attribute__((ext_vector_type(8)))  float    v8f;

#define N_AT  16000
#define E_B   34000
#define E_S   256000
#define N_RT  1200
#define E_T   28800
#define CDIM  192   // NS + 3*NV
#define NSC   48
#define HD    64

__device__ __forceinline__ v8f wmma_f16(v16h a, v16h b, v8f c) {
  return __builtin_amdgcn_wmma_f32_16x16x32_f16(false, a, false, b, (short)0, c, false, false);
}
// A-matrix per-lane K mapping (16-bit A 16x32, ISA 7.12.2): lane = m + 16h
__device__ __forceinline__ int amap(int t, int h) { return 8*h + t + ((t >= 8) ? 8 : 0); }

// B fragment (K x N row-major f16 in LDS): lane = n + 16h, slot t -> K = kc*32+amap(t,h)
__device__ __forceinline__ v16h ldsBfrag(const f16* W, int ldw, int n, int h, int kc, int n0) {
  v16h b;
#pragma unroll
  for (int t = 0; t < 16; ++t) b[t] = W[(kc*32 + amap(t, h))*ldw + n0 + n];
  return b;
}
// B fragment straight from global f32 weights (L2-resident)
__device__ __forceinline__ v16h glbBfrag(const float* W, int ldw, int n, int h, int kc, int n0) {
  v16h b;
#pragma unroll
  for (int t = 0; t < 16; ++t) b[t] = (f16)W[(kc*32 + amap(t, h))*ldw + n0 + n];
  return b;
}
// A fragment from a row-major 16 x ld f16 tile in LDS
__device__ __forceinline__ v16h ldsAfrag(const f16* S, int ld, int m, int h, int kc) {
  v16h a;
#pragma unroll
  for (int t = 0; t < 16; ++t) a[t] = S[m*ld + kc*32 + amap(t, h)];
  return a;
}

// ------------------------------ utility ------------------------------------
__global__ void k_zero(float* __restrict__ p, long n) {
  long i = (long)blockIdx.x * blockDim.x + threadIdx.x;
  if (i < n) p[i] = 0.f;
}

// ------------------- fused tensor-conv edge kernel -------------------------
// One wave == one 16-edge tile. WMMA MLP (64->64 relu ->240), then the
// tensor product runs directly from the WMMA accumulator fragments
// (lane n holds channels {n,n+16,n+32} of all five 48-wide w blocks).
__global__ __launch_bounds__(128) void k_conv_edge(
    const float* __restrict__ feat, const float* __restrict__ ef,
    const float* __restrict__ sh,   const int*   __restrict__ eidx,
    const float* __restrict__ w1g,  const float* __restrict__ b1g,
    const float* __restrict__ w2g,  const float* __restrict__ b2g,
    float* __restrict__ accum, float* __restrict__ cnt, int E)
{
  __shared__ f16 sW1[64*64];
  __shared__ f16 sW2[64*240];
  __shared__ float sB2[240];
  __shared__ f16 sH[4][16*64];

  const int tid = threadIdx.x;
  for (int i = tid; i < 64*64;  i += 128) sW1[i] = (f16)w1g[i];
  for (int i = tid; i < 64*240; i += 128) sW2[i] = (f16)w2g[i];
  for (int i = tid; i < 240;    i += 128) sB2[i] = b2g[i];
  __syncthreads();

  const int wv = tid >> 5, lane = tid & 31;
  const int l15 = lane & 15, h = lane >> 4;
  const long e0 = ((long)blockIdx.x * 4 + wv) * 16;
  const long em_ = e0 + l15;
  const int  em  = (int)(em_ < E ? em_ : (E - 1));

  // A fragments: this tile's edge features (K = 64 -> 2 chunks)
  v16h a0, a1;
  {
    const float* er = ef + (size_t)em * 64;
#pragma unroll
    for (int t = 0; t < 16; ++t) { int k = amap(t, h); a0[t] = (f16)er[k]; a1[t] = (f16)er[32 + k]; }
  }
  f16* sh1 = sH[wv];
  // GEMM1 + bias + relu, store to LDS in A-ready layout
#pragma unroll
  for (int j = 0; j < 4; ++j) {
    v8f acc = {};
    acc = wmma_f16(a0, ldsBfrag(sW1, 64, l15, h, 0, j*16), acc);
    acc = wmma_f16(a1, ldsBfrag(sW1, 64, l15, h, 1, j*16), acc);
    int col = j*16 + l15; float bb = b1g[col];
#pragma unroll
    for (int r = 0; r < 8; ++r) { float x = acc[r] + bb; sh1[(r + 8*h)*64 + col] = (f16)(x > 0.f ? x : 0.f); }
  }
  __syncthreads();
  v16h h0f = ldsAfrag(sh1, 64, l15, h, 0);
  v16h h1f = ldsAfrag(sh1, 64, l15, h, 1);
  // GEMM2: (16x64) @ (64x240) -> 15 N-tiles kept in registers
  v8f accw[15];
#pragma unroll
  for (int j = 0; j < 15; ++j) {
    v8f acc = {};
    acc = wmma_f16(h0f, ldsBfrag(sW2, 240, l15, h, 0, j*16), acc);
    acc = wmma_f16(h1f, ldsBfrag(sW2, 240, l15, h, 1, j*16), acc);
    accw[j] = acc;
  }
  // tensor product + atomic scatter-mean numerator
#pragma unroll
  for (int r = 0; r < 8; ++r) {
    const int row = r + 8*h;
    const long er_ = e0 + row; const bool valid = er_ < E;
    const long er = valid ? er_ : 0;
    const int dst = eidx[er], src = eidx[(long)E + er];
    const float* shr = sh + (size_t)er * 9;
    const float y0 = shr[0], y1x = shr[1], y1y = shr[2], y1z = shr[3];
    const float a1c = shr[4], a2c = shr[5], a3c = shr[6], a4c = shr[7], a5c = shr[8];
    const float* fr = feat + (size_t)dst * CDIM;
    float* ar = accum + (size_t)src * CDIM;
#pragma unroll
    for (int q = 0; q < 3; ++q) {
      const int c = l15 + q*16;      // channel; w block p lives in N-tile 3p+q, lane col l15
      float wv5[5];
#pragma unroll
      for (int p = 0; p < 5; ++p) wv5[p] = accw[3*p + q][r] + sB2[48*p + c];
      const float s  = fr[c];
      const float vx = fr[48 + 3*c + 0], vy = fr[48 + 3*c + 1], vz = fr[48 + 3*c + 2];
      const float vd = vx*y1x + vy*y1y + vz*y1z;
      const float Mx = a1c*vx + a2c*vy + a3c*vz;
      const float My = a2c*vx + a4c*vy + a5c*vz;
      const float Mz = a3c*vx + a5c*vy + (-a1c - a4c)*vz;
      const float os  = wv5[0]*s*y0 + wv5[1]*vd;
      const float ovx = wv5[2]*s*y1x + wv5[3]*y0*vx + wv5[4]*Mx;
      const float ovy = wv5[2]*s*y1y + wv5[3]*y0*vy + wv5[4]*My;
      const float ovz = wv5[2]*s*y1z + wv5[3]*y0*vz + wv5[4]*Mz;
      if (valid) {
        atomicAdd(ar + c, os);
        atomicAdd(ar + 48 + 3*c + 0, ovx);
        atomicAdd(ar + 48 + 3*c + 1, ovy);
        atomicAdd(ar + 48 + 3*c + 2, ovz);
      }
    }
    if (valid && l15 == 0) atomicAdd(cnt + src, 1.f);
  }
}

// ------------------- node: scatter-mean + residual + BN stats --------------
__global__ __launch_bounds__(192) void k_node_statsA(
    float* __restrict__ accum, const float* __restrict__ feat,
    const float* __restrict__ cnt, float* __restrict__ stats, int N)
{
  const int c = threadIdx.x;           // channel 0..191
  const int n0 = blockIdx.x * 64;
  float s1 = 0.f, s2 = 0.f;
  for (int i = 0; i < 64; ++i) {
    const int n = n0 + i; if (n >= N) break;
    float cc = cnt[n]; cc = cc > 1.f ? cc : 1.f;
    const long idx = (long)n * CDIM + c;
    const float o = accum[idx] / cc + feat[idx];
    accum[idx] = o;
    if (c < 48) { s1 += o; s2 += o*o; } else { s2 += o*o; }
  }
  if (c < 48) { atomicAdd(stats + c, s1); atomicAdd(stats + 48 + c, s2); }
  else          atomicAdd(stats + 96 + (c - 48)/3, s2 * (1.f/3.f));
}

__global__ void k_node_fin(float* __restrict__ stats, int N) {
  const int c = threadIdx.x;           // 96 threads
  const float inv = 1.f / (float)N;
  if (c < 48) {
    const float mean = stats[c] * inv;
    const float var  = stats[48 + c] * inv - mean*mean;
    stats[c] = mean; stats[48 + c] = rsqrtf(var + 1e-5f);
  } else {
    const int v = c - 48;
    stats[96 + v] = rsqrtf(stats[96 + v] * inv + 1e-5f);
  }
}

__global__ void k_node_apply(
    const float* __restrict__ accum, float* __restrict__ feat,
    const float* __restrict__ stats, const float* __restrict__ gs,
    const float* __restrict__ bs, const float* __restrict__ gv, long total)
{
  const long i = (long)blockIdx.x * blockDim.x + threadIdx.x;
  if (i >= total) return;
  const int c = (int)(i % CDIM);
  const float o = accum[i];
  float f;
  if (c < 48) f = (o - stats[c]) * stats[48 + c] * gs[c] + bs[c];
  else { const int v = (c - 48)/3; f = o * gv[v] * stats[96 + v]; }
  feat[i] = f;
}

// a = out[:, :48] @ eu_lin (48x64)
__global__ void k_node_lin(const float* __restrict__ feat, const float* __restrict__ lin,
                           float* __restrict__ abuf, int N) {
  const int idx = blockIdx.x * blockDim.x + threadIdx.x;
  if (idx >= N * 64) return;
  const int n = idx >> 6, k = idx & 63;
  const float* fr = feat + (size_t)n * CDIM;
  float s = 0.f;
  for (int c = 0; c < 48; ++c) s += fr[c] * lin[c*64 + k];
  abuf[idx] = s;
}

// ---------------- fused edge-update MLP (bond layers) ----------------------
__global__ __launch_bounds__(128) void k_edge_update(
    const float* __restrict__ abuf, float* __restrict__ efb,
    const int* __restrict__ eidx,
    const float* __restrict__ w1g, const float* __restrict__ b1g,
    const float* __restrict__ w2g, const float* __restrict__ b2g,
    const float* __restrict__ w3g, const float* __restrict__ b3g,
    const float* __restrict__ lng, const float* __restrict__ lnb, int E)
{
  __shared__ f16 sW1[192*64];
  __shared__ f16 sW2[64*64];
  __shared__ f16 sW3[64*64];
  __shared__ f16 sH[4][16*64];
  const int tid = threadIdx.x;
  for (int i = tid; i < 192*64; i += 128) sW1[i] = (f16)w1g[i];
  for (int i = tid; i < 64*64;  i += 128) { sW2[i] = (f16)w2g[i]; sW3[i] = (f16)w3g[i]; }
  __syncthreads();

  const int wv = tid >> 5, lane = tid & 31;
  const int l15 = lane & 15, h = lane >> 4;
  const long e0 = ((long)blockIdx.x * 4 + wv) * 16;
  const long em_ = e0 + l15;
  const int  em  = (int)(em_ < E ? em_ : (E - 1));
  const int dstm = eidx[em], srcm = eidx[(long)E + em];
  const float* pa = abuf + (size_t)dstm * 64;
  const float* pb = abuf + (size_t)srcm * 64;
  const float* pe = efb  + (size_t)em   * 64;

  v16h A[6];  // ein = [a[dst] | a[src] | ef], K = 192
#pragma unroll
  for (int kc = 0; kc < 6; ++kc)
#pragma unroll
    for (int t = 0; t < 16; ++t) {
      const int K = kc*32 + amap(t, h);
      const float x = (K < 64) ? pa[K] : (K < 128) ? pb[K - 64] : pe[K - 128];
      A[kc][t] = (f16)x;
    }
  f16* sh1 = sH[wv];
#pragma unroll
  for (int j = 0; j < 4; ++j) {      // GEMM1: 192 -> 64
    v8f acc = {};
#pragma unroll
    for (int kc = 0; kc < 6; ++kc) acc = wmma_f16(A[kc], ldsBfrag(sW1, 64, l15, h, kc, j*16), acc);
    int col = j*16 + l15; float bb = b1g[col];
#pragma unroll
    for (int r = 0; r < 8; ++r) { float x = acc[r] + bb; sh1[(r + 8*h)*64 + col] = (f16)(x > 0.f ? x : 0.f); }
  }
  __syncthreads();
  v16h B0 = ldsAfrag(sh1, 64, l15, h, 0), B1 = ldsAfrag(sh1, 64, l15, h, 1);
  __syncthreads();
#pragma unroll
  for (int j = 0; j < 4; ++j) {      // GEMM2: 64 -> 64
    v8f acc = {};
    acc = wmma_f16(B0, ldsBfrag(sW2, 64, l15, h, 0, j*16), acc);
    acc = wmma_f16(B1, ldsBfrag(sW2, 64, l15, h, 1, j*16), acc);
    int col = j*16 + l15; float bb = b2g[col];
#pragma unroll
    for (int r = 0; r < 8; ++r) { float x = acc[r] + bb; sh1[(r + 8*h)*64 + col] = (f16)(x > 0.f ? x : 0.f); }
  }
  __syncthreads();
  v16h C0 = ldsAfrag(sh1, 64, l15, h, 0), C1 = ldsAfrag(sh1, 64, l15, h, 1);
  v8f acc3[4];
#pragma unroll
  for (int j = 0; j < 4; ++j) {      // GEMM3: 64 -> 64 (no relu)
    v8f acc = {};
    acc = wmma_f16(C0, ldsBfrag(sW3, 64, l15, h, 0, j*16), acc);
    acc = wmma_f16(C1, ldsBfrag(sW3, 64, l15, h, 1, j*16), acc);
    acc3[j] = acc;
  }
  // residual + LayerNorm over the 64 cols of each edge row
#pragma unroll
  for (int r = 0; r < 8; ++r) {
    const int row = r + 8*h;
    const long er_ = e0 + row; const bool valid = er_ < E;
    const long er = valid ? er_ : 0;
    const float* efr = efb + (size_t)er * 64;
    float ev[4]; float s1 = 0.f, s2 = 0.f;
#pragma unroll
    for (int j = 0; j < 4; ++j) {
      const int col = j*16 + l15;
      const float x = acc3[j][r] + b3g[col] + efr[col];
      ev[j] = x; s1 += x; s2 += x*x;
    }
#pragma unroll
    for (int msk = 1; msk < 16; msk <<= 1) { s1 += __shfl_xor(s1, msk, 16); s2 += __shfl_xor(s2, msk, 16); }
    const float mu = s1 * (1.f/64.f);
    const float var = s2 * (1.f/64.f) - mu*mu;
    const float rstd = rsqrtf(var + 1e-5f);
    if (valid) {
#pragma unroll
      for (int j = 0; j < 4; ++j) {
        const int col = j*16 + l15;
        efb[(size_t)er * 64 + col] = (ev[j] - mu) * rstd * lng[col] + lnb[col];
      }
    }
  }
}

// ---------------------------- torsion head ---------------------------------
__global__ void k_rot(const float* __restrict__ pos, const int* __restrict__ bei,
                      const int* __restrict__ rb, float* __restrict__ bpos,
                      float* __restrict__ axisv, int* __restrict__ rot1, int NR, int EB)
{
  const int r = blockIdx.x * blockDim.x + threadIdx.x; if (r >= NR) return;
  const int e = rb[r]; const int a0 = bei[e], a1 = bei[EB + e];
  const float p0x = pos[a0*3], p0y = pos[a0*3+1], p0z = pos[a0*3+2];
  const float p1x = pos[a1*3], p1y = pos[a1*3+1], p1z = pos[a1*3+2];
  bpos[r*3+0] = 0.5f*(p0x + p1x); bpos[r*3+1] = 0.5f*(p0y + p1y); bpos[r*3+2] = 0.5f*(p0z + p1z);
  float ax = p1x - p0x, ay = p1y - p0y, az = p1z - p0z;
  const float nn = sqrtf(ax*ax + ay*ay + az*az) + 1e-8f;
  axisv[r*3+0] = ax/nn; axisv[r*3+1] = ay/nn; axisv[r*3+2] = az/nn;
  rot1[r] = a1;
}

// fused: RBF -> emb MLP (32->64->64) -> tor MLP (160->160 relu ->192) -> TP -> scatter
__global__ __launch_bounds__(64) void k_tor_edge(
    const float* __restrict__ feat, const float* __restrict__ pos,
    const float* __restrict__ bpos, const int* __restrict__ tei,
    const float* __restrict__ ew1, const float* __restrict__ eb1,
    const float* __restrict__ ew2, const float* __restrict__ eb2,
    const float* __restrict__ fw1, const float* __restrict__ fb1,
    const float* __restrict__ fw2, const float* __restrict__ fb2,
    float* __restrict__ accT, float* __restrict__ cntT, int ET)
{
  __shared__ f16 sE[2][16*64];
  __shared__ f16 sH1[2][16*160];
  __shared__ f16 sR[2][16*32];
  __shared__ int sTa[2][16], sJ[2][16];
  __shared__ float sD[2][16];
  __shared__ float sU[2][64];

  const int tid = threadIdx.x; const int wv = tid >> 5, lane = tid & 31;
  const int l15 = lane & 15, h = lane >> 4;
  const long e0 = ((long)blockIdx.x * 2 + wv) * 16;
  if (lane < 16) {
    const long e_ = e0 + lane; const int e = (int)(e_ < ET ? e_ : (ET - 1));
    const int j  = tei[e];
    const int ta = tei[(long)ET + e];
    const int jb = tei[j];                 // reference quirk: bond idx used as atom idx
    const int ja = tei[(long)ET + j];
    sTa[wv][lane] = ta; sJ[wv][lane] = j;
    const float ex = pos[ta*3+0] - bpos[j*3+0];
    const float ey = pos[ta*3+1] - bpos[j*3+1];
    const float ez = pos[ta*3+2] - bpos[j*3+2];
    sD[wv][lane] = sqrtf(ex*ex + ey*ey + ez*ez);
    float rx = pos[ja*3+0] - pos[jb*3+0];
    float ry = pos[ja*3+1] - pos[jb*3+1];
    float rz = pos[ja*3+2] - pos[jb*3+2];
    const float rn = sqrtf(rx*rx + ry*ry + rz*rz) + 1e-8f;
    rx /= rn; ry /= rn; rz /= rn;
    const float s75 = 2.7386127875258306f;   // sqrt(7.5)
    sU[wv][lane*4+0] = s75*(rx*rx - (1.f/3.f));
    sU[wv][lane*4+1] = s75*rx*ry;
    sU[wv][lane*4+2] = s75*rx*rz;
    sU[wv][lane*4+3] = s75*(ry*ry - (1.f/3.f));
  }
  __syncthreads();
  for (int i = lane; i < 512; i += 32) {   // RBF(32) per edge
    const int mrow = i >> 5, k = i & 31;
    const float d = sD[wv][mrow];
    const float mu = 5.f * (float)k / 31.f;
    const float t = (d - mu) * 6.4f;       // / sig, sig = 5/32
    sR[wv][i] = (f16)__expf(-t*t);
  }
  __syncthreads();
  f16* sEw = sE[wv];
  v16h ra = ldsAfrag(sR[wv], 32, l15, h, 0);
#pragma unroll
  for (int j = 0; j < 4; ++j) {            // emb GEMM1: 32 -> 64, relu
    v8f acc = {};
    acc = wmma_f16(ra, glbBfrag(ew1, 64, l15, h, 0, j*16), acc);
    int col = j*16 + l15; float bb = eb1[col];
#pragma unroll
    for (int r = 0; r < 8; ++r) { float x = acc[r] + bb; sEw[(r + 8*h)*64 + col] = (f16)(x > 0.f ? x : 0.f); }
  }
  __syncthreads();
  v16h ea0 = ldsAfrag(sEw, 64, l15, h, 0), ea1 = ldsAfrag(sEw, 64, l15, h, 1);
  __syncthreads();
#pragma unroll
  for (int j = 0; j < 4; ++j) {            // emb GEMM2: 64 -> 64 (linear)
    v8f acc = {};
    acc = wmma_f16(ea0, glbBfrag(ew2, 64, l15, h, 0, j*16), acc);
    acc = wmma_f16(ea1, glbBfrag(ew2, 64, l15, h, 1, j*16), acc);
    int col = j*16 + l15; float bb = eb2[col];
#pragma unroll
    for (int r = 0; r < 8; ++r) sEw[(r + 8*h)*64 + col] = (f16)(acc[r] + bb);
  }
  __syncthreads();
  // X = [e_attr(64) | feat[ta,:48] | tba(48)], K = 160
  v16h A[5];
  {
    const int ta = sTa[wv][l15]; const int j = sJ[wv][l15];
    const int jb = tei[j], ja = tei[(long)ET + j];
    const float* fta = feat + (size_t)ta * CDIM;
    const float* fjb = feat + (size_t)jb * CDIM;
    const float* fja = feat + (size_t)ja * CDIM;
#pragma unroll
    for (int kc = 0; kc < 5; ++kc)
#pragma unroll
      for (int t = 0; t < 16; ++t) {
        const int K = kc*32 + amap(t, h);
        float x;
        if (K < 64)       x = (float)sEw[l15*64 + K];
        else if (K < 112) x = fta[K - 64];
        else              x = fjb[K - 112] + fja[K - 112];
        A[kc][t] = (f16)x;
      }
  }
  f16* sh1 = sH1[wv];
#pragma unroll
  for (int j = 0; j < 10; ++j) {           // tor GEMM1: 160 -> 160, relu
    v8f acc = {};
#pragma unroll
    for (int kc = 0; kc < 5; ++kc) acc = wmma_f16(A[kc], glbBfrag(fw1, 160, l15, h, kc, j*16), acc);
    int col = j*16 + l15; float bb = fb1[col];
#pragma unroll
    for (int r = 0; r < 8; ++r) { float x = acc[r] + bb; sh1[(r + 8*h)*160 + col] = (f16)(x > 0.f ? x : 0.f); }
  }
  __syncthreads();
  v16h A2[5];
#pragma unroll
  for (int kc = 0; kc < 5; ++kc) A2[kc] = ldsAfrag(sh1, 160, l15, h, kc);
  v8f accw[12];
#pragma unroll
  for (int j = 0; j < 12; ++j) {           // tor GEMM2: 160 -> 192
    v8f acc = {};
#pragma unroll
    for (int kc = 0; kc < 5; ++kc) acc = wmma_f16(A2[kc], glbBfrag(fw2, 192, l15, h, kc, j*16), acc);
    accw[j] = acc;
  }
  // tensor product + scatter by rot-bond index
#pragma unroll
  for (int r = 0; r < 8; ++r) {
    const int row = r + 8*h;
    const long er_ = e0 + row; const bool valid = er_ < ET;
    const int ta = sTa[wv][row], j = sJ[wv][row];
    const float u0 = sU[wv][row*4+0], u1 = sU[wv][row*4+1], u2 = sU[wv][row*4+2], cc = sU[wv][row*4+3];
    const float* fr = feat + (size_t)ta * CDIM;
    float* ar = accT + (size_t)j * 96;
#pragma unroll
    for (int q = 0; q < 3; ++q) {
      const int c = l15 + q*16;
      const float w0 = accw[q]     [r] + fb2[c];
      const float w1 = accw[3 + q] [r] + fb2[48 + c];
      const float w2 = accw[6 + q] [r] + fb2[96 + c];
      const float w3 = accw[9 + q] [r] + fb2[144 + c];
      const float s  = fr[c];
      const float vx = fr[48 + 3*c + 0], vy = fr[48 + 3*c + 1], vz = fr[48 + 3*c + 2];
      const float vd = vx*u0 + vy*u1 + vz*u2;
      if (valid) {
        atomicAdd(ar + c,      w0*s*cc + w1*vd);
        atomicAdd(ar + 48 + c, w2*s*cc + w3*vd);
      }
    }
    if (valid && l15 == 0) atomicAdd(cntT + j, 1.f);
  }
}

__global__ __launch_bounds__(96) void k_torA(
    float* __restrict__ accT, const float* __restrict__ cntT,
    float* __restrict__ statsT, int NR)
{
  const int c = threadIdx.x;
  const int r0 = blockIdx.x * 64;
  float s1 = 0.f, s2 = 0.f;
  for (int i = 0; i < 64; ++i) {
    const int r = r0 + i; if (r >= NR) break;
    float cc = cntT[r]; cc = cc > 1.f ? cc : 1.f;
    const float o = accT[(long)r*96 + c] / cc;
    accT[(long)r*96 + c] = o;
    s1 += o; s2 += o*o;
  }
  atomicAdd(statsT + c, s1); atomicAdd(statsT + 96 + c, s2);
}

__global__ void k_torFin(float* __restrict__ statsT, int NR) {
  const int c = threadIdx.x;               // 96 threads
  const float inv = 1.f / (float)NR;
  const float mean = statsT[c] * inv;
  const float var  = statsT[96 + c] * inv - mean*mean;
  statsT[c] = mean; statsT[96 + c] = rsqrtf(var + 1e-5f);
}

__global__ void k_torB(
    const float* __restrict__ accT, const float* __restrict__ statsT,
    const float* __restrict__ g, const float* __restrict__ b,
    const float* __restrict__ w1, const float* __restrict__ w2,
    float* __restrict__ torb, float* __restrict__ outTor, int NR)
{
  const int r = blockIdx.x * blockDim.x + threadIdx.x; if (r >= NR) return;
  float on[96];
  for (int c = 0; c < 96; ++c)
    on[c] = (accT[(long)r*96 + c] - statsT[c]) * statsT[96 + c] * g[c] + b[c];
  float t = 0.f;
  for (int k = 0; k < 48; ++k) {
    float y = 0.f;
    for (int c = 0; c < 96; ++c) y += on[c] * w1[c*48 + k];
    y = y > 0.f ? y : 0.f;
    t += y * w2[k];
  }
  const float val = tanhf(t) * 3.14159265358979323846f + 1e-4f;
  torb[r] = val; outTor[r] = val;
}

__global__ void k_newpos(
    const float* __restrict__ pos, const int* __restrict__ ui,
    const float* __restrict__ axisv, const int* __restrict__ rot1,
    const float* __restrict__ torb, float* __restrict__ outp, int N)
{
  const int i = blockIdx.x * blockDim.x + threadIdx.x; if (i >= N) return;
  const int b = ui[i];
  const float kx = axisv[b*3], ky = axisv[b*3+1], kz = axisv[b*3+2];
  const int pa = rot1[b];
  const float px = pos[pa*3], py = pos[pa*3+1], pz = pos[pa*3+2];
  const float ang = torb[b];
  const float prx = pos[i*3]   - px;
  const float pry = pos[i*3+1] - py;
  const float prz = pos[i*3+2] - pz;
  const float cv = cosf(ang), sv = sinf(ang), om = 1.f - cv;
  const float crx = ky*prz - kz*pry;
  const float cry = kz*prx - kx*prz;
  const float crz = kx*pry - ky*prx;
  const float kd = kx*prx + ky*pry + kz*prz;
  outp[i*3+0] = px + prx*cv + crx*sv + kx*kd*om;
  outp[i*3+1] = py + pry*cv + cry*sv + ky*kd*om;
  outp[i*3+2] = pz + prz*cv + crz*sv + kz*kd*om;
}

// ---------------------------------------------------------------------------
#define CDIV(a,b) (((a)+(b)-1)/(b))

extern "C" void kernel_launch(void* const* d_in, const int* in_sizes, int n_in,
                              void* d_out, int out_size, void* d_ws, size_t ws_size,
                              hipStream_t stream) {
  (void)in_sizes; (void)n_in; (void)out_size; (void)ws_size;
  const float* atom_features = (const float*)d_in[0];
  const float* atom_pos      = (const float*)d_in[1];
  const float* bond_features = (const float*)d_in[2];
  const float* bond_sh       = (const float*)d_in[3];
  const int*   bond_ei       = (const int*)  d_in[4];
  const float* sp_features   = (const float*)d_in[5];
  const float* sp_sh         = (const float*)d_in[6];
  const int*   sp_ei         = (const int*)  d_in[7];
  const int*   rb            = (const int*)  d_in[8];
  const int*   ui            = (const int*)  d_in[10];
  const int*   tei           = (const int*)  d_in[11];
  auto P = [&](int i) { return (const float*)d_in[12 + i]; };

  // workspace layout (floats)
  float* ws = (float*)d_ws; size_t o = 0;
  float* feat   = ws + o; o += (size_t)N_AT * CDIM;       // 3,072,000
  float* accum  = ws + o; o += (size_t)N_AT * CDIM;       // 3,072,000 (+cnt+stats contiguous for zeroing)
  float* cnt    = ws + o; o += N_AT;                      // 16,000
  float* stats  = ws + o; o += 256;
  float* efb    = ws + o; o += (size_t)E_B * HD;          // 2,176,000
  float* abuf   = ws + o; o += (size_t)N_AT * HD;         // 1,024,000
  float* bpos   = ws + o; o += N_RT * 3;
  float* axisv  = ws + o; o += N_RT * 3;
  int*   rot1   = (int*)(ws + o); o += N_RT;
  float* accT   = ws + o; o += (size_t)N_RT * 96;         // 115,200 (+cntT+statsT contiguous)
  float* cntT   = ws + o; o += N_RT;
  float* statsT = ws + o; o += 256;
  float* torb   = ws + o; o += N_RT;

  hipMemcpyAsync(feat, atom_features, (size_t)N_AT * CDIM * 4, hipMemcpyDeviceToDevice, stream);
  hipMemcpyAsync(efb,  bond_features, (size_t)E_B  * HD   * 4, hipMemcpyDeviceToDevice, stream);

  const long zconv = (long)N_AT * CDIM + N_AT + 256;      // accum|cnt|stats
  const int  gbB = CDIV(CDIV(E_B, 16), 4);                // bond edge-tile blocks (4 waves)
  const int  gbS = CDIV(CDIV(E_S, 16), 4);

  for (int L = 0; L < 5; ++L) {
    const int pb = L * 16;          // bond param base
    const int ps = 80 + L * 7;      // spatial param base
    // ---- bond conv (update_edge = true) ----
    k_zero<<<CDIV(zconv, 256), 256, 0, stream>>>(accum, zconv);
    k_conv_edge<<<gbB, 128, 0, stream>>>(feat, efb, bond_sh, bond_ei,
        P(pb+0), P(pb+1), P(pb+2), P(pb+3), accum, cnt, E_B);
    k_node_statsA<<<CDIV(N_AT, 64), 192, 0, stream>>>(accum, feat, cnt, stats, N_AT);
    k_node_fin<<<1, 96, 0, stream>>>(stats, N_AT);
    k_node_apply<<<CDIV((long)N_AT*CDIM, 256), 256, 0, stream>>>(accum, feat, stats,
        P(pb+4), P(pb+5), P(pb+6), (long)N_AT * CDIM);
    k_node_lin<<<CDIV(N_AT*64, 256), 256, 0, stream>>>(feat, P(pb+7), abuf, N_AT);
    k_edge_update<<<gbB, 128, 0, stream>>>(abuf, efb, bond_ei,
        P(pb+8), P(pb+9), P(pb+10), P(pb+11), P(pb+12), P(pb+13), P(pb+14), P(pb+15), E_B);
    // ---- spatial conv (update_edge = false) ----
    k_zero<<<CDIV(zconv, 256), 256, 0, stream>>>(accum, zconv);
    k_conv_edge<<<gbS, 128, 0, stream>>>(feat, sp_features, sp_sh, sp_ei,
        P(ps+0), P(ps+1), P(ps+2), P(ps+3), accum, cnt, E_S);
    k_node_statsA<<<CDIV(N_AT, 64), 192, 0, stream>>>(accum, feat, cnt, stats, N_AT);
    k_node_fin<<<1, 96, 0, stream>>>(stats, N_AT);
    k_node_apply<<<CDIV((long)N_AT*CDIM, 256), 256, 0, stream>>>(accum, feat, stats,
        P(ps+4), P(ps+5), P(ps+6), (long)N_AT * CDIM);
  }

  // ---- torsion head ----
  const long ztor = (long)N_RT * 96 + N_RT + 256;
  k_zero<<<CDIV(ztor, 256), 256, 0, stream>>>(accT, ztor);
  k_rot<<<CDIV(N_RT, 128), 128, 0, stream>>>(atom_pos, bond_ei, rb, bpos, axisv, rot1, N_RT, E_B);
  k_tor_edge<<<CDIV(CDIV(E_T, 16), 2), 64, 0, stream>>>(feat, atom_pos, bpos, tei,
      P(115), P(116), P(117), P(118),     // emb_w1, emb_b1, emb_w2, emb_b2
      P(119), P(120), P(121), P(122),     // tor fc_w1, fc_b1, fc_w2, fc_b2
      accT, cntT, E_T);
  k_torA<<<CDIV(N_RT, 64), 96, 0, stream>>>(accT, cntT, statsT, N_RT);
  k_torFin<<<1, 96, 0, stream>>>(statsT, N_RT);

  float* out = (float*)d_out;
  k_torB<<<CDIV(N_RT, 128), 128, 0, stream>>>(accT, statsT,
      P(123), P(124), P(125), P(126), torb, out + 5296000, N_RT);
  k_newpos<<<CDIV(N_AT, 256), 256, 0, stream>>>(atom_pos, ui, axisv, rot1, torb, out, N_AT);

  hipMemcpyAsync(out + 48000,   feat, (size_t)N_AT * CDIM * 4, hipMemcpyDeviceToDevice, stream);
  hipMemcpyAsync(out + 3120000, efb,  (size_t)E_B  * HD   * 4, hipMemcpyDeviceToDevice, stream);
}
